// SemanticAwareQuerySampler_32272384262297
// MI455X (gfx1250) — matrix operand order repaired and checked by reading, not compiled, not run on hardware
//
#include <hip/hip_runtime.h>
#include <math.h>

// ---------------------------------------------------------------------------
// CDNA5 (gfx1250) WMMA types
// ---------------------------------------------------------------------------
typedef __bf16 bf16_t;
typedef bf16_t v16bf __attribute__((ext_vector_type(16)));
typedef float  v8f   __attribute__((ext_vector_type(8)));

union ABu {
    v16bf    v;
    unsigned u[8];
};

#define B_      16
#define MAXQ    800
#define HD_     80     // density H
#define WD_     80     // density W
#define E_      256    // embed dim
#define NH_     8      // heads
#define HDIM_   32     // head dim
#define NKEY_   256    // agent tokens
#define NBQ     (B_ * MAXQ)       // 12800
#define HW_     (HD_ * WD_)       // 6400

// f32 -> bf16 round-to-nearest-even
__device__ __forceinline__ unsigned short f2bf(float x) {
    unsigned u = __float_as_uint(x);
    u = (u + 0x7FFFu + ((u >> 16) & 1u)) >> 16;
    return (unsigned short)u;
}
__device__ __forceinline__ int clampq(float c) {
    int n = (int)floorf(c * 2.0f);                 // ALPHA = 2.0
    return n < 100 ? 100 : (n > 800 ? 800 : n);
}
__device__ __forceinline__ unsigned hashu(unsigned x) {
    x ^= x >> 16; x *= 0x7feb352dU;
    x ^= x >> 15; x *= 0x846ca68bU;
    x ^= x >> 16;
    return x;
}

// ---------------------------------------------------------------------------
// Kernel 0: nq (as float) and padding_mask outputs
// ---------------------------------------------------------------------------
__global__ void head_kernel(const float* __restrict__ pcount,
                            float* __restrict__ out_nq,
                            float* __restrict__ out_mask) {
    int t = blockIdx.x * blockDim.x + threadIdx.x;
    if (t < B_) out_nq[t] = (float)clampq(pcount[t]);
    if (t < NBQ) {
        int b = t / MAXQ, q = t % MAXQ;
        out_mask[t] = (q >= clampq(pcount[b])) ? 1.0f : 0.0f;
    }
}

// ---------------------------------------------------------------------------
// Kernel 1a: f32 -> bf16 conversion (already (N,K) "torch" weights, acts)
// ---------------------------------------------------------------------------
__global__ void cvt_bf16_kernel(const float* __restrict__ s,
                                unsigned short* __restrict__ d, int n) {
    int i = blockIdx.x * blockDim.x + threadIdx.x;
    if (i < n) d[i] = f2bf(s[i]);
}

// ---------------------------------------------------------------------------
// Kernel 1b: f32 (K,N) -> bf16 (N,K) transpose-convert, so ALL GEMMs use the
// coalesced dword-load weight path (no scalar u16 gather path).
// ---------------------------------------------------------------------------
__global__ void cvt_bf16_T_kernel(const float* __restrict__ s,  // (Kd,Nd)
                                  unsigned short* __restrict__ d, // (Nd,Kd)
                                  int Kd, int Nd) {
    int i = blockIdx.x * blockDim.x + threadIdx.x;
    if (i < Kd * Nd) {
        int k = i / Nd, n = i % Nd;
        d[(size_t)n * Kd + k] = f2bf(s[i]);
    }
}

// ---------------------------------------------------------------------------
// Kernel 2: Gumbel-max categorical sampling. One wave32 per (b,q).
// ---------------------------------------------------------------------------
__global__ __launch_bounds__(32) void sample_kernel(
        const float* __restrict__ density,
        const float* __restrict__ pcount,
        float* __restrict__ positions) {
    int bq = blockIdx.x;
    int b = bq / MAXQ, q = bq % MAXQ;
    int lane = threadIdx.x;

    int nq = clampq(pcount[b]);
    const float invT = 10.0f;   // 1 / TEMPERATURE

    float best = -3.0e38f;
    int   bidx = 0;
    unsigned salt = (unsigned)bq * 0x9E3779B9u ^ 0xB5297A4Du;
    for (int i = lane; i < HW_; i += 32) {
        unsigned h = hashu((unsigned)i * 2654435761u ^ salt);
        float u = (float)(h >> 8) * (1.0f / 16777216.0f) + 5.96e-8f;
        float g = -__logf(-__logf(u));
        float val = density[b * HW_ + i] * invT + g;
        if (val > best) { best = val; bidx = i; }
    }
#pragma unroll
    for (int off = 16; off >= 1; off >>= 1) {
        float ov = __shfl_xor(best, off, 32);
        int   oi = __shfl_xor(bidx, off, 32);
        if (ov > best || (ov == best && oi < bidx)) { best = ov; bidx = oi; }
    }
    if (lane == 0) {
        float px = 0.0f, py = 0.0f;
        if (q < nq) {
            py = (float)(bidx / WD_) * (1.0f / (float)HD_);
            px = (float)(bidx % WD_) * (1.0f / (float)WD_);
        }
        positions[bq * 2 + 0] = px;
        positions[bq * 2 + 1] = py;
    }
}

// ---------------------------------------------------------------------------
// Kernel 3: pos MLP layer 1 (2 -> 128) + tanh-GELU, bf16 output
// ---------------------------------------------------------------------------
__global__ void pos1_kernel(const float* __restrict__ positions,
                            const float* __restrict__ w1,   // (2,128)
                            const float* __restrict__ b1,   // (128)
                            unsigned short* __restrict__ H) {
    int bq = blockIdx.x, j = threadIdx.x;   // 128 threads
    float x = positions[bq * 2 + 0];
    float y = positions[bq * 2 + 1];
    float t = x * w1[j] + y * w1[128 + j] + b1[j];
    float g = 0.5f * t * (1.0f + tanhf(0.7978845608f * (t + 0.044715f * t * t * t)));
    H[(size_t)bq * 128 + j] = f2bf(g);
}

// ---------------------------------------------------------------------------
// Kernel 4: bilinear gather of encoder memory at sampled positions
// ---------------------------------------------------------------------------
__global__ void bilinear_kernel(const float* __restrict__ positions,
                                const float* __restrict__ mem,   // (B,8400,256)
                                unsigned short* __restrict__ S) {
    int bq = blockIdx.x, c = threadIdx.x;    // 256 threads
    int b = bq / MAXQ;
    float xf = positions[bq * 2 + 0] * (float)WD_ - 0.5f;
    float yf = positions[bq * 2 + 1] * (float)HD_ - 0.5f;
    int x0 = (int)floorf(xf), y0 = (int)floorf(yf);
    float wx1 = xf - (float)x0, wx0 = 1.0f - wx1;
    float wy1 = yf - (float)y0, wy0 = 1.0f - wy1;

    const float* fm = mem + (size_t)b * 8400 * E_;
    auto g = [&](int yy, int xx) -> float {
        if (yy < 0 || yy >= HD_ || xx < 0 || xx >= WD_) return 0.0f;
        return fm[(size_t)(yy * WD_ + xx) * E_ + c];
    };
    float v = g(y0,     x0    ) * (wy0 * wx0)
            + g(y0,     x0 + 1) * (wy0 * wx1)
            + g(y0 + 1, x0    ) * (wy1 * wx0)
            + g(y0 + 1, x0 + 1) * (wy1 * wx1);
    S[(size_t)bq * E_ + c] = f2bf(v);
}

// ---------------------------------------------------------------------------
// Kernel 5: bf16 WMMA GEMM, 16(M) x 64(N) tile per wave (4 accumulators).
// W is always (N,K) row-major -> B[k][n] = W[n*K+k] (coalesced dword loads).
// A layout  (ISA 16-bit A 16x32):  lane->M, VGPR v holds K pair
//   k = kc + (v<4?0:16) + 8*(lane>>4) + 2*(v&3)
// B layout  (ISA 16-bit B KxN):    lane->N, VGPR v holds K pair
//   k = kc + (v<4?0:16) + 8*(lane>>4) + 2*(v&3)
// A operand is loaded once per k-chunk and reused for all 4 N-tiles.
// ---------------------------------------------------------------------------
__global__ __launch_bounds__(32) void gemm_bf16_wmma(
        const unsigned short* __restrict__ A,
        const unsigned short* __restrict__ W,   // (N,K)
        int K,
        const float* __restrict__ bias,
        const float* __restrict__ addsrc,
        float* __restrict__ outF,
        unsigned short* __restrict__ outB,
        int N) {
    int lane = threadIdx.x;
    int lm = lane & 15, half = lane >> 4;
    int m0 = blockIdx.y * 16;
    int n0 = blockIdx.x * 64;

    v8f acc[4] = {};
    const unsigned short* arow  = A + (size_t)(m0 + lm) * K;
    const unsigned short* wrow0 = W + (size_t)(n0 + lm) * K;

    for (int kc = 0; kc < K; kc += 32) {
        ABu a;
#pragma unroll
        for (int v = 0; v < 8; ++v) {
            int k = kc + ((v < 4) ? 0 : 16) + 8 * half + 2 * (v & 3);
            a.u[v] = *(const unsigned*)(arow + k);
        }
#pragma unroll
        for (int t = 0; t < 4; ++t) {
            ABu bm;
            const unsigned short* wrow = wrow0 + (size_t)t * 16 * K;
#pragma unroll
            for (int v = 0; v < 8; ++v) {
                int k = kc + ((v < 4) ? 0 : 16) + 8 * half + 2 * (v & 3);
                bm.u[v] = *(const unsigned*)(wrow + k);
            }
            acc[t] = __builtin_amdgcn_wmma_f32_16x16x32_bf16(
                        false, a.v, false, bm.v, (short)0, acc[t], false, false);
        }
    }

#pragma unroll
    for (int t = 0; t < 4; ++t) {
        int n = n0 + t * 16 + lm;
        float bn = bias ? bias[n] : 0.0f;
#pragma unroll
        for (int r = 0; r < 8; ++r) {
            int m = m0 + r + 8 * half;
            size_t oi = (size_t)m * N + n;
            float vv = acc[t][r] + bn;
            if (addsrc) vv += addsrc[oi];
            if (outF) outF[oi] = vv;
            if (outB) outB[oi] = f2bf(vv);
        }
    }
}

// ---------------------------------------------------------------------------
// Kernel 6: fused attention for one (b, head, 16-query tile) per wave32.
//   scores(16x256) via 16 WMMA (K = head_dim = 32), in-register softmax
//   (half-wave shfl reductions), P staged to LDS (bf16), context(16x32)
//   via 16 WMMA (2 N-tiles x 8 K-chunks).
// ---------------------------------------------------------------------------
__global__ __launch_bounds__(32) void attn_kernel(
        const unsigned short* __restrict__ Q,    // (B,800,256) bf16
        const unsigned short* __restrict__ Kb,   // (B,256,256) bf16
        const unsigned short* __restrict__ Vb,   // (B,256,256) bf16
        unsigned short* __restrict__ Ctx) {      // (B,800,256) bf16
    int blk = blockIdx.x;
    int mt = blk % (MAXQ / 16);
    int h  = (blk / (MAXQ / 16)) % NH_;
    int b  = blk / ((MAXQ / 16) * NH_);
    int lane = threadIdx.x;
    int lm = lane & 15, half = lane >> 4;
    int m0 = mt * 16;

    __shared__ unsigned short P[16 * NKEY_];   // 8 KB

    // ---- load Q A-operand (16 x 32, head slice) ----
    ABu qa;
    const unsigned short* qrow = Q + ((size_t)(b * MAXQ + m0 + lm) * E_) + h * HDIM_;
#pragma unroll
    for (int v = 0; v < 8; ++v) {
        int k = ((v < 4) ? 0 : 16) + 8 * half + 2 * (v & 3);
        qa.u[v] = *(const unsigned*)(qrow + k);
    }

    // ---- scores: one WMMA per 16-key tile ----
    v8f sc[16];
#pragma unroll
    for (int j = 0; j < 16; ++j) {
        ABu kb;
        const unsigned short* krow =
            Kb + ((size_t)(b * NKEY_ + j * 16 + lm) * E_) + h * HDIM_;
#pragma unroll
        for (int v = 0; v < 8; ++v) {
            int k = ((v < 4) ? 0 : 16) + 8 * half + 2 * (v & 3);
            kb.u[v] = *(const unsigned*)(krow + k);
        }
        v8f z = {};
        sc[j] = __builtin_amdgcn_wmma_f32_16x16x32_bf16(
                    false, qa.v, false, kb.v, (short)0, z, false, false);
    }

    // ---- softmax over 256 keys per row (rows live across half-wave lanes) ----
    const float scale = 0.1767766952966369f;   // 1/sqrt(32)
#pragma unroll
    for (int r = 0; r < 8; ++r) {
        float mval = -3.0e38f;
#pragma unroll
        for (int j = 0; j < 16; ++j) mval = fmaxf(mval, sc[j][r]);
#pragma unroll
        for (int off = 8; off >= 1; off >>= 1)
            mval = fmaxf(mval, __shfl_xor(mval, off, 32));
        float s = 0.0f;
#pragma unroll
        for (int j = 0; j < 16; ++j) {
            float e = __expf((sc[j][r] - mval) * scale);
            sc[j][r] = e;
            s += e;
        }
#pragma unroll
        for (int off = 8; off >= 1; off >>= 1)
            s += __shfl_xor(s, off, 32);
        float inv = 1.0f / s;
#pragma unroll
        for (int j = 0; j < 16; ++j) {
            float p = sc[j][r] * inv;
            P[(r + 8 * half) * NKEY_ + j * 16 + lm] = f2bf(p);
        }
    }
    __syncthreads();

    // ---- context = P(16x256) * V(256x32): 2 N-tiles x 8 K-chunks ----
#pragma unroll
    for (int nt = 0; nt < 2; ++nt) {
        v8f o = {};
#pragma unroll
        for (int kc = 0; kc < 8; ++kc) {
            ABu pa, vb;
#pragma unroll
            for (int v = 0; v < 8; ++v) {
                int k = kc * 32 + ((v < 4) ? 0 : 16) + 8 * half + 2 * (v & 3);
                pa.u[v] = *(const unsigned*)&P[lm * NKEY_ + k];
            }
#pragma unroll
            for (int v = 0; v < 8; ++v) {
                int k = kc * 32 + ((v < 4) ? 0 : 16) + 8 * half + 2 * (v & 3);
                unsigned short lo =
                    Vb[((size_t)(b * NKEY_ + k) * E_) + h * HDIM_ + nt * 16 + lm];
                unsigned short hi =
                    Vb[((size_t)(b * NKEY_ + k + 1) * E_) + h * HDIM_ + nt * 16 + lm];
                vb.u[v] = ((unsigned)hi << 16) | (unsigned)lo;
            }
            o = __builtin_amdgcn_wmma_f32_16x16x32_bf16(
                    false, pa.v, false, vb.v, (short)0, o, false, false);
        }
#pragma unroll
        for (int r = 0; r < 8; ++r) {
            int m = m0 + r + 8 * half;
            Ctx[((size_t)(b * MAXQ + m) * E_) + h * HDIM_ + nt * 16 + lm] = f2bf(o[r]);
        }
    }
}

// ---------------------------------------------------------------------------
// Host-side launcher
// ---------------------------------------------------------------------------
extern "C" void kernel_launch(void* const* d_in, const int* in_sizes, int n_in,
                              void* d_out, int out_size, void* d_ws, size_t ws_size,
                              hipStream_t stream) {
    const float* density = (const float*)d_in[0];
    const float* pcount  = (const float*)d_in[1];
    const float* mem     = (const float*)d_in[2];
    const float* agent   = (const float*)d_in[3];
    const float* pos_w1  = (const float*)d_in[4];
    const float* pos_b1  = (const float*)d_in[5];
    const float* pos_w2  = (const float*)d_in[6];
    const float* pos_b2  = (const float*)d_in[7];
    const float* in_w    = (const float*)d_in[8];
    const float* in_b    = (const float*)d_in[9];
    const float* out_w   = (const float*)d_in[10];
    const float* out_b   = (const float*)d_in[11];
    const float* qi_w    = (const float*)d_in[12];
    const float* qi_b    = (const float*)d_in[13];

    float* outQ  = (float*)d_out;                       // queries   (B,800,256)
    float* outQP = outQ + (size_t)NBQ * E_;             // query_pos (B,800,256)
    float* outNQ = outQP + (size_t)NBQ * E_;            // nq (B,)
    float* outPM = outNQ + B_;                          // padding_mask (B,800)

    size_t off = 0;
    auto alloc = [&](size_t bytes) -> void* {
        void* p = (char*)d_ws + off;
        off += (bytes + 255) & ~(size_t)255;
        return p;
    };
    float*          posbuf  = (float*)alloc((size_t)NBQ * 2 * 4);
    unsigned short* inw_bf  = (unsigned short*)alloc((size_t)768 * 256 * 2);  // (N,K) rows
    unsigned short* outw_bf = (unsigned short*)alloc((size_t)256 * 256 * 2);  // (N,K)
    unsigned short* w2T_bf  = (unsigned short*)alloc((size_t)256 * 128 * 2);  // (N=256,K=128)
    unsigned short* qiT_bf  = (unsigned short*)alloc((size_t)256 * 64 * 2);   // (N=256,K=64)
    unsigned short* ag_bf   = (unsigned short*)alloc((size_t)B_ * 256 * 64 * 2);
    unsigned short* agi_bf  = (unsigned short*)alloc((size_t)B_ * 256 * 256 * 2);
    unsigned short* smp_bf  = (unsigned short*)alloc((size_t)NBQ * 256 * 2);
    unsigned short* h_bf    = (unsigned short*)alloc((size_t)NBQ * 128 * 2);
    unsigned short* q_bf    = (unsigned short*)alloc((size_t)NBQ * 256 * 2);
    unsigned short* k_bf    = (unsigned short*)alloc((size_t)B_ * 256 * 256 * 2);
    unsigned short* v_bf    = (unsigned short*)alloc((size_t)B_ * 256 * 256 * 2);
    unsigned short* ctx_bf  = (unsigned short*)alloc((size_t)NBQ * 256 * 2);

    // 0) nq + padding mask outputs
    head_kernel<<<(NBQ + 255) / 256, 256, 0, stream>>>(pcount, outNQ, outPM);

    // 1) weight/activation bf16 conversions ((K,N) weights transposed to (N,K))
    cvt_bf16_kernel<<<(768 * 256 + 255) / 256, 256, 0, stream>>>(in_w,  inw_bf,  768 * 256);
    cvt_bf16_kernel<<<(256 * 256 + 255) / 256, 256, 0, stream>>>(out_w, outw_bf, 256 * 256);
    cvt_bf16_T_kernel<<<(128 * 256 + 255) / 256, 256, 0, stream>>>(pos_w2, w2T_bf, 128, 256);
    cvt_bf16_T_kernel<<<(64 * 256 + 255) / 256, 256, 0, stream>>>(qi_w,  qiT_bf, 64, 256);
    cvt_bf16_kernel<<<(B_ * 256 * 64 + 255) / 256, 256, 0, stream>>>(agent, ag_bf, B_ * 256 * 64);

    // 2) Gumbel-max categorical sampling (wave per query)
    sample_kernel<<<NBQ, 32, 0, stream>>>(density, pcount, posbuf);

    // 3) pos MLP layer 1 + GELU
    pos1_kernel<<<NBQ, 128, 0, stream>>>(posbuf, pos_w1, pos_b1, h_bf);

    // 4) bilinear feature gather
    bilinear_kernel<<<NBQ, 256, 0, stream>>>(posbuf, mem, smp_bf);

    // 5) agent_init = agent @ qinit_w + b       (4096 x 64) x (64 x 256)
    gemm_bf16_wmma<<<dim3(256 / 64, (B_ * 256) / 16), 32, 0, stream>>>(
        ag_bf, qiT_bf, 64, qi_b, nullptr, nullptr, agi_bf, 256);

    // 6) Q = sampled @ Wq^T + bq                (12800 x 256) x (256 x 256)
    gemm_bf16_wmma<<<dim3(256 / 64, NBQ / 16), 32, 0, stream>>>(
        smp_bf, inw_bf, 256, in_b, nullptr, nullptr, q_bf, 256);

    // 7) K = agent_init @ Wk^T + bk
    gemm_bf16_wmma<<<dim3(256 / 64, (B_ * 256) / 16), 32, 0, stream>>>(
        agi_bf, inw_bf + (size_t)256 * 256, 256, in_b + 256, nullptr, nullptr, k_bf, 256);

    // 8) V = agent_init @ Wv^T + bv
    gemm_bf16_wmma<<<dim3(256 / 64, (B_ * 256) / 16), 32, 0, stream>>>(
        agi_bf, inw_bf + (size_t)512 * 256, 256, in_b + 512, nullptr, nullptr, v_bf, 256);

    // 9) query_pos = h @ pos_w2 + b2 -> f32 output slot 1
    gemm_bf16_wmma<<<dim3(256 / 64, NBQ / 16), 32, 0, stream>>>(
        h_bf, w2T_bf, 128, pos_b2, nullptr, outQP, nullptr, 256);

    // 10) fused multi-head attention -> context (bf16)
    attn_kernel<<<B_ * NH_ * (MAXQ / 16), 32, 0, stream>>>(q_bf, k_bf, v_bf, ctx_bf);

    // 11) queries = ctx @ Wout^T + bo + query_pos -> f32 output slot 0
    gemm_bf16_wmma<<<dim3(256 / 64, NBQ / 16), 32, 0, stream>>>(
        ctx_bf, outw_bf, 256, out_b, outQP, outQ, nullptr, 256);

    (void)in_sizes; (void)n_in; (void)out_size; (void)ws_size;
}
